// MultisourcePerceiverCrossAttention_74371653697605
// MI455X (gfx1250) — matrix operand order repaired and checked by reading, not compiled, not run on hardware
//
#include <hip/hip_runtime.h>
#include <hip/hip_fp16.h>

typedef __attribute__((ext_vector_type(16))) _Float16 v16h;
typedef __attribute__((ext_vector_type(8)))  _Float16 v8h;
typedef __attribute__((ext_vector_type(8)))  float    v8f;

#define B_   4
#define N_   512
#define M_   8192
#define DV_  512
#define DC_  128
#define H_   8
#define DI_  512
#define HD_  64
#define BH_  (B_*H_)

#define GEMM_F32     0
#define GEMM_RMS_KB  1   // rms-norm rows, write logits-B-fragment layout (K tensors)
#define GEMM_VB      2   // write PV-B-fragment layout (V tensors)
#define GEMM_RMS_A   3   // rms-norm rows, write A-fragment layout (Q tensors)

__device__ __forceinline__ v8f wmma_f16(v16h a, v16h b, v8f c) {
  return __builtin_amdgcn_wmma_f32_16x16x32_f16(false, a, false, b, (short)0, c,
                                                false, false);
}

// ---------------------------------------------------------------------------
// Swizzle [R,K] row-major into WMMA A-fragment-ready blocks:
// frag f = rtile*(K/32)+kc, 512 f16 per frag, lane-major (lane*16 + elem).
// A layout (16x32 f16): lane l: row = l&15, half = l>>4;
//   elem e <-> K_local = (e&7) + 8*half + 16*(e>>3)
// ---------------------------------------------------------------------------
template <typename T>
__global__ void swizzleA_kernel(const T* __restrict__ in, _Float16* __restrict__ out,
                                int R, int K) {
  const int kcN = K >> 5;
  const size_t total = (size_t)R * K;
  for (size_t idx = (size_t)blockIdx.x * blockDim.x + threadIdx.x; idx < total;
       idx += (size_t)gridDim.x * blockDim.x) {
    size_t f = idx >> 9;
    int l = (int)((idx >> 4) & 31);
    int e = (int)(idx & 15);
    int rt = (int)(f / kcN), kc = (int)(f % kcN);
    int row  = rt * 16 + (l & 15);
    int half = l >> 4;
    int kl   = (e & 7) + (half << 3) + ((e >> 3) << 4);
    out[idx] = (_Float16)(float)in[(size_t)row * K + (kc * 32 + kl)];
  }
}

// ---------------------------------------------------------------------------
// Swizzle weights [K,N] row-major into WMMA B-fragment-ready blocks:
// frag f = ntile*(K/32)+kc. B layout (32x16 f16): lane l: N = l&15,
//   elem e <-> K_local = (e&7) + 8*(l>>4) + 16*(e>>3)
// ---------------------------------------------------------------------------
__global__ void swizzleB_kernel(const float* __restrict__ in, _Float16* __restrict__ out,
                                int K, int Ncols) {
  const int kcN = K >> 5;
  const size_t total = (size_t)K * Ncols;
  for (size_t idx = (size_t)blockIdx.x * blockDim.x + threadIdx.x; idx < total;
       idx += (size_t)gridDim.x * blockDim.x) {
    size_t f = idx >> 9;
    int l = (int)((idx >> 4) & 31);
    int e = (int)(idx & 15);
    int nt = (int)(f / kcN), kc = (int)(f % kcN);
    int k = kc * 32 + (e & 7) + ((l >> 4) << 3) + ((e >> 3) << 4);
    int n = nt * 16 + (l & 15);
    out[idx] = (_Float16)in[(size_t)k * Ncols + n];
  }
}

// ---------------------------------------------------------------------------
// Fused GEMM: out[16-row strip, Ncols] = A[strip,:K] @ W[K,Ncols] + bias,
// optional rowwise RMS-norm * gain, then epilogue re-swizzle into the
// fragment layout the attention kernel consumes.
// Block = 256 threads (8 waves); Ncols == TPW*8*16, wave w owns ntiles
// w, w+8, ... (TPW of them) -- compile-time TPW keeps the K-loop branch-free
// so acc stays pinned and the body is pure loads + v_wmma.
// ---------------------------------------------------------------------------
template <int TPW>
__global__ __launch_bounds__(256) void gemm16_fused_kernel(
    const _Float16* __restrict__ Asw, const _Float16* __restrict__ Bsw,
    const float* __restrict__ bias, const float* __restrict__ gain,
    float* __restrict__ outf, _Float16* __restrict__ outh,
    int K, int mode, int seg) {
  constexpr int Ncols  = TPW * 128;
  __shared__ float strip[16][Ncols];
  __shared__ float red[16][16];
  __shared__ float rmsf[16];

  const int rt   = blockIdx.x;
  const int tid  = threadIdx.x;
  const int wave = tid >> 5, lane = tid & 31;
  const int kcN  = K >> 5;

  v8f acc[TPW];
#pragma unroll
  for (int t = 0; t < TPW; ++t)
#pragma unroll
    for (int i = 0; i < 8; ++i) acc[t][i] = 0.f;

  const _Float16* Ab = Asw + ((size_t)rt * kcN) * 512 + (size_t)lane * 16;
  const _Float16* Bb = Bsw + (size_t)lane * 16;
#pragma unroll 1
  for (int kc = 0; kc < kcN; ++kc) {
    v16h a = *(const v16h*)(Ab + (size_t)kc * 512);
#pragma unroll
    for (int t = 0; t < TPW; ++t) {
      const int nt = wave + (t << 3);
      v16h bf = *(const v16h*)(Bb + ((size_t)nt * kcN + kc) * 512);
      acc[t] = wmma_f16(a, bf, acc[t]);
    }
  }

  // C/D layout: lane l, elem i -> row = i + 8*(l>>4), col-in-tile = l&15
  const int cl = lane & 15, hf = lane >> 4;
#pragma unroll
  for (int t = 0; t < TPW; ++t) {
    const int nt = wave + (t << 3);
    const int c = nt * 16 + cl;
    float bv = bias ? bias[c] : 0.f;
#pragma unroll
    for (int i = 0; i < 8; ++i) strip[i + 8 * hf][c] = acc[t][i] + bv;
  }
  __syncthreads();

  if (mode == GEMM_RMS_KB || mode == GEMM_RMS_A) {
    int r = tid & 15, ck = tid >> 4;
    constexpr int cpc = Ncols >> 4;
    float p = 0.f;
#pragma unroll
    for (int c = 0; c < cpc; ++c) {
      float v = strip[r][ck * cpc + c];
      p += v * v;
    }
    red[r][ck] = p;
    __syncthreads();
    if (tid < 16) {
      float s = 0.f;
#pragma unroll
      for (int j = 0; j < 16; ++j) s += red[tid][j];
      rmsf[tid] = rsqrtf(s / (float)Ncols + 1e-6f);
    }
    __syncthreads();
  }

  const int tilesPerSeg = seg >> 4;
  const int bidx = rt / tilesPerSeg;
  const int mt   = rt - bidx * tilesPerSeg;
  constexpr int nelem = 16 * Ncols;
  for (int j = tid; j < nelem; j += 256) {
    int r = j / Ncols, c = j % Ncols;
    float v = strip[r][c];
    if (mode == GEMM_F32) {
      outf[((size_t)rt * 16 + r) * Ncols + c] = v;
      continue;
    }
    int h = c >> 6, hd = c & 63;
    int bh = bidx * H_ + h;
    if (mode == GEMM_RMS_KB) {
      v *= rmsf[r] * gain[c];
      int kcd = hd >> 5, hd5 = hd & 31;
      int ld = r + (((hd5 & 15) >> 3) << 4);
      int e  = (hd5 & 7) | ((hd5 >> 4) << 3);
      size_t frag = ((size_t)bh * tilesPerSeg + mt) * 2 + kcd;
      outh[frag * 512 + (size_t)ld * 16 + e] = (_Float16)v;
    } else if (mode == GEMM_VB) {
      int nt4 = hd >> 4, hdl = hd & 15;
      int mt32 = mt >> 1, par = mt & 1;
      int ld = hdl + (par << 4);
      size_t frag = ((size_t)bh * (tilesPerSeg >> 1) + mt32) * 4 + nt4;
      outh[frag * 512 + (size_t)ld * 16 + r] = (_Float16)v;
    } else {  // GEMM_RMS_A
      v *= rmsf[r] * gain[c];
      int kcd = hd >> 5, hd5 = hd & 31;
      int ld = r + (((hd5 & 15) >> 3) << 4);
      int e  = (hd5 & 7) | ((hd5 >> 4) << 3);
      size_t frag = ((size_t)bh * tilesPerSeg + mt) * 2 + kcd;
      outh[frag * 512 + (size_t)ld * 16 + e] = (_Float16)v;
    }
  }
}

// ---------------------------------------------------------------------------
// Flash attention: one wave per 16-query tile of one (b,h).
// logits = (qc@kc^T + lambda*qv@kv^T)/8, online softmax over M (8192 keys in
// 32-key chunks), fused accumulation of 3 value streams. Row sums of P are
// computed with one extra WMMA against an all-ones B fragment (replaces a
// 4-step ds_bpermute ladder per row) and land exactly in lrun[] indexing.
// ---------------------------------------------------------------------------
__global__ __launch_bounds__(128) void flash_attention_kernel(
    const _Float16* __restrict__ qc_sw, const _Float16* __restrict__ qv_sw,
    const _Float16* __restrict__ kc_sw, const _Float16* __restrict__ kv_sw,
    const _Float16* __restrict__ v_sw, const float* __restrict__ lam_p,
    _Float16* __restrict__ attn_out) {
  __shared__ __align__(16) _Float16 plds[4][16][32];
  const int wave = threadIdx.x >> 5, lane = threadIdx.x & 31;
  const int qt = blockIdx.x * 4 + wave;
  const int bh = qt >> 5;  // N_/16 == 32 query tiles per (b,h)
  const int nt = qt & 31;
  const int b = bh >> 3, h = bh & 7;
  const float lam = lam_p[0];
  const int cl = lane & 15, hf = lane >> 4;

  v16h aqc[2], aqv[2];
  {
    size_t qb = (((size_t)bh * (N_ / 16) + nt) * 2) * 512 + (size_t)lane * 16;
    aqc[0] = *(const v16h*)(qc_sw + qb);
    aqc[1] = *(const v16h*)(qc_sw + qb + 512);
    aqv[0] = *(const v16h*)(qv_sw + qb);
    aqv[1] = *(const v16h*)(qv_sw + qb + 512);
  }

  v16h onesB;
#pragma unroll
  for (int e = 0; e < 16; ++e) onesB[e] = (_Float16)1.0f;

  v8f acc[3][4];
#pragma unroll
  for (int s = 0; s < 3; ++s)
#pragma unroll
    for (int t = 0; t < 4; ++t)
#pragma unroll
      for (int i = 0; i < 8; ++i) acc[s][t][i] = 0.f;
  float mrun[8], lrun[8];
#pragma unroll
  for (int i = 0; i < 8; ++i) { mrun[i] = -3.0e38f; lrun[i] = 0.f; }

  const size_t vstride = (size_t)BH_ * (M_ / 32) * 4 * 512;

#pragma unroll 1
  for (int ch = 0; ch < M_ / 32; ++ch) {
    // ---- dual-source logits for two 16-key subtiles ----
    v8f sl[2];
#pragma unroll
    for (int st = 0; st < 2; ++st) {
      size_t kb = (((size_t)bh * (M_ / 16) + (ch * 2 + st)) * 2) * 512 + (size_t)lane * 16;
      v8f cc, cv;
#pragma unroll
      for (int i = 0; i < 8; ++i) { cc[i] = 0.f; cv[i] = 0.f; }
      cc = wmma_f16(aqc[0], *(const v16h*)(kc_sw + kb), cc);
      cc = wmma_f16(aqc[1], *(const v16h*)(kc_sw + kb + 512), cc);
      cv = wmma_f16(aqv[0], *(const v16h*)(kv_sw + kb), cv);
      cv = wmma_f16(aqv[1], *(const v16h*)(kv_sw + kb + 512), cv);
#pragma unroll
      for (int i = 0; i < 8; ++i) sl[st][i] = (cc[i] + lam * cv[i]) * 0.125f;
    }

    // ---- online max update (rows live in 16-lane halves under wave32) ----
    float scale[8];
#pragma unroll
    for (int i = 0; i < 8; ++i) {
      float rm = fmaxf(sl[0][i], sl[1][i]);
#pragma unroll
      for (int m = 1; m < 16; m <<= 1) rm = fmaxf(rm, __shfl_xor(rm, m, 32));
      float mn = fmaxf(mrun[i], rm);
      scale[i] = __expf(mrun[i] - mn);
      mrun[i] = mn;
      sl[0][i] = __expf(sl[0][i] - mn);
      sl[1][i] = __expf(sl[1][i] - mn);
    }
#pragma unroll
    for (int s = 0; s < 3; ++s)
#pragma unroll
      for (int t = 0; t < 4; ++t)
#pragma unroll
        for (int i = 0; i < 8; ++i) acc[s][t][i] *= scale[i];

    // ---- C-layout -> A-layout transpose of P through per-wave LDS ----
#pragma unroll
    for (int st = 0; st < 2; ++st)
#pragma unroll
      for (int i = 0; i < 8; ++i)
        plds[wave][i + 8 * hf][st * 16 + cl] = (_Float16)sl[st][i];
    // same-wave LDS ops are in-order; compiler inserts s_wait_dscnt for RAW
    v8h lo = *(const v8h*)&plds[wave][cl][16 * hf];
    v8h hi = *(const v8h*)&plds[wave][cl][16 * hf + 8];
    v16h pA;
#pragma unroll
    for (int e = 0; e < 8; ++e) { pA[e] = lo[e]; pA[e + 8] = hi[e]; }

    // ---- row sums of P via WMMA with all-ones B: D[i] = sum of row i+8h ----
    {
      v8f ps;
#pragma unroll
      for (int i = 0; i < 8; ++i) ps[i] = 0.f;
      ps = wmma_f16(pA, onesB, ps);
#pragma unroll
      for (int i = 0; i < 8; ++i) lrun[i] = lrun[i] * scale[i] + ps[i];
    }

    // ---- P @ V for the 3 value streams ----
    const _Float16* vb =
        v_sw + (((size_t)bh * (M_ / 32) + ch) * 4) * 512 + (size_t)lane * 16;
#pragma unroll
    for (int s = 0; s < 3; ++s) {
      const _Float16* vp = vb + (size_t)s * vstride;
#pragma unroll
      for (int t = 0; t < 4; ++t)
        acc[s][t] = wmma_f16(pA, *(const v16h*)(vp + (size_t)t * 512), acc[s][t]);
    }

    if (ch + 1 < M_ / 32) {  // global_prefetch_b8 for next chunk
      size_t kb2 = (((size_t)bh * (M_ / 16) + (ch * 2 + 2)) * 2) * 512;
      __builtin_prefetch(kc_sw + kb2, 0, 1);
      __builtin_prefetch(kv_sw + kb2, 0, 1);
      __builtin_prefetch(vb + 4 * 512, 0, 1);
    }
  }

  // ---- normalize and write merged-head output [3][B][N][DI] (f16) ----
#pragma unroll
  for (int s = 0; s < 3; ++s)
#pragma unroll
    for (int t = 0; t < 4; ++t)
#pragma unroll
      for (int i = 0; i < 8; ++i) {
        int n = nt * 16 + i + 8 * hf;
        int c = h * 64 + t * 16 + cl;
        attn_out[(((size_t)s * B_ + b) * N_ + n) * DI_ + c] =
            (_Float16)(acc[s][t][i] / lrun[i]);
      }
}

// ---------------------------------------------------------------------------
extern "C" void kernel_launch(void* const* d_in, const int* in_sizes, int n_in,
                              void* d_out, int out_size, void* d_ws, size_t ws_size,
                              hipStream_t stream) {
  (void)in_sizes; (void)n_in; (void)out_size; (void)ws_size;
  const float* V   = (const float*)d_in[0];
  const float* C   = (const float*)d_in[1];
  const float* D   = (const float*)d_in[2];
  const float* Lv  = (const float*)d_in[3];
  const float* Lc  = (const float*)d_in[4];
  const float* lam = (const float*)d_in[5];
  const float* Wqv = (const float*)d_in[6];  const float* bqv = (const float*)d_in[7];  const float* gqv = (const float*)d_in[8];
  const float* Wqc = (const float*)d_in[9];  const float* bqc = (const float*)d_in[10]; const float* gqc = (const float*)d_in[11];
  const float* Wkv = (const float*)d_in[12]; const float* bkv = (const float*)d_in[13]; const float* gkv = (const float*)d_in[14];
  const float* Wkc = (const float*)d_in[15]; const float* bkc = (const float*)d_in[16]; const float* gkc = (const float*)d_in[17];
  const float* Wvv = (const float*)d_in[18]; const float* bvv = (const float*)d_in[19];
  const float* Wvc = (const float*)d_in[20]; const float* bvc = (const float*)d_in[21];
  const float* Wvd = (const float*)d_in[22]; const float* bvd = (const float*)d_in[23];
  const float* Wov = (const float*)d_in[24]; const float* bov = (const float*)d_in[25];
  const float* Woc = (const float*)d_in[26]; const float* boc = (const float*)d_in[27];
  const float* Wod = (const float*)d_in[28]; const float* bod = (const float*)d_in[29];
  float* out = (float*)d_out;

  const int RKV = B_ * M_;  // 32768
  const int RQ  = B_ * N_;  // 2048

  char* ws = (char*)d_ws;
  size_t off = 0;
  auto alloc_h = [&](size_t nelem) -> _Float16* {
    _Float16* p = (_Float16*)(ws + off);
    off = (off + nelem * sizeof(_Float16) + 255) & ~(size_t)255;
    return p;
  };

  _Float16* Vsw  = alloc_h((size_t)RKV * DV_);
  _Float16* Csw  = alloc_h((size_t)RKV * DC_);
  _Float16* Dsw  = alloc_h((size_t)RKV * DV_);
  _Float16* Lvsw = alloc_h((size_t)RQ * DV_);
  _Float16* Lcsw = alloc_h((size_t)RQ * DC_);
  _Float16* Wqv_s = alloc_h((size_t)DV_ * DI_);
  _Float16* Wqc_s = alloc_h((size_t)DC_ * DI_);
  _Float16* Wkv_s = alloc_h((size_t)DV_ * DI_);
  _Float16* Wkc_s = alloc_h((size_t)DC_ * DI_);
  _Float16* Wvv_s = alloc_h((size_t)DV_ * DI_);
  _Float16* Wvc_s = alloc_h((size_t)DC_ * DI_);
  _Float16* Wvd_s = alloc_h((size_t)DV_ * DI_);
  _Float16* Wov_s = alloc_h((size_t)DI_ * DV_);
  _Float16* Woc_s = alloc_h((size_t)DI_ * DC_);
  _Float16* Wod_s = alloc_h((size_t)DI_ * DV_);
  _Float16* qc_sw = alloc_h((size_t)RQ * DI_);
  _Float16* qv_sw = alloc_h((size_t)RQ * DI_);
  _Float16* kc_sw = alloc_h((size_t)RKV * DI_);
  _Float16* kv_sw = alloc_h((size_t)RKV * DI_);
  _Float16* v_sw  = alloc_h((size_t)3 * RKV * DI_);
  _Float16* attn  = alloc_h((size_t)3 * RQ * DI_);
  _Float16* attn_sw = alloc_h((size_t)3 * RQ * DI_);

  // --- stage 0: cast + swizzle GEMM row operands and weights into frag layout
  swizzleA_kernel<float><<<2048, 256, 0, stream>>>(V,  Vsw,  RKV, DV_);
  swizzleA_kernel<float><<<1024, 256, 0, stream>>>(C,  Csw,  RKV, DC_);
  swizzleA_kernel<float><<<2048, 256, 0, stream>>>(D,  Dsw,  RKV, DV_);
  swizzleA_kernel<float><<<256,  256, 0, stream>>>(Lv, Lvsw, RQ,  DV_);
  swizzleA_kernel<float><<<128,  256, 0, stream>>>(Lc, Lcsw, RQ,  DC_);
  swizzleB_kernel<<<256, 256, 0, stream>>>(Wqv, Wqv_s, DV_, DI_);
  swizzleB_kernel<<<64,  256, 0, stream>>>(Wqc, Wqc_s, DC_, DI_);
  swizzleB_kernel<<<256, 256, 0, stream>>>(Wkv, Wkv_s, DV_, DI_);
  swizzleB_kernel<<<64,  256, 0, stream>>>(Wkc, Wkc_s, DC_, DI_);
  swizzleB_kernel<<<256, 256, 0, stream>>>(Wvv, Wvv_s, DV_, DI_);
  swizzleB_kernel<<<64,  256, 0, stream>>>(Wvc, Wvc_s, DC_, DI_);
  swizzleB_kernel<<<256, 256, 0, stream>>>(Wvd, Wvd_s, DV_, DI_);
  swizzleB_kernel<<<256, 256, 0, stream>>>(Wov, Wov_s, DI_, DV_);
  swizzleB_kernel<<<64,  256, 0, stream>>>(Woc, Woc_s, DI_, DC_);
  swizzleB_kernel<<<256, 256, 0, stream>>>(Wod, Wod_s, DI_, DV_);

  // --- stage 1: projections (WMMA GEMM + bias [+ RMS] -> attention layouts)
  gemm16_fused_kernel<4><<<RQ / 16, 256, 0, stream>>>(Lvsw, Wqv_s, bqv, gqv, nullptr, qv_sw, DV_, GEMM_RMS_A, N_);
  gemm16_fused_kernel<4><<<RQ / 16, 256, 0, stream>>>(Lcsw, Wqc_s, bqc, gqc, nullptr, qc_sw, DC_, GEMM_RMS_A, N_);
  gemm16_fused_kernel<4><<<RKV / 16, 256, 0, stream>>>(Vsw, Wkv_s, bkv, gkv, nullptr, kv_sw, DV_, GEMM_RMS_KB, M_);
  gemm16_fused_kernel<4><<<RKV / 16, 256, 0, stream>>>(Csw, Wkc_s, bkc, gkc, nullptr, kc_sw, DC_, GEMM_RMS_KB, M_);
  const size_t vstride = (size_t)RKV * DI_;
  gemm16_fused_kernel<4><<<RKV / 16, 256, 0, stream>>>(Vsw, Wvv_s, bvv, nullptr, nullptr, v_sw + 0 * vstride, DV_, GEMM_VB, M_);
  gemm16_fused_kernel<4><<<RKV / 16, 256, 0, stream>>>(Csw, Wvc_s, bvc, nullptr, nullptr, v_sw + 1 * vstride, DC_, GEMM_VB, M_);
  gemm16_fused_kernel<4><<<RKV / 16, 256, 0, stream>>>(Dsw, Wvd_s, bvd, nullptr, nullptr, v_sw + 2 * vstride, DV_, GEMM_VB, M_);

  // --- stage 2: fused dual-QK flash attention, 3 value streams
  flash_attention_kernel<<<(BH_ * (N_ / 16)) / 4, 128, 0, stream>>>(
      qc_sw, qv_sw, kc_sw, kv_sw, v_sw, lam, attn);

  // --- stage 3: output projections
  swizzleA_kernel<_Float16><<<1024, 256, 0, stream>>>(attn, attn_sw, 3 * RQ, DI_);
  gemm16_fused_kernel<4><<<RQ / 16, 256, 0, stream>>>(
      attn_sw, Wov_s, bov, nullptr, out, nullptr, DI_, GEMM_F32, N_);
  gemm16_fused_kernel<1><<<RQ / 16, 256, 0, stream>>>(
      attn_sw + (size_t)RQ * DI_, Woc_s, boc, nullptr, out + (size_t)RQ * DV_,
      nullptr, DI_, GEMM_F32, N_);
  gemm16_fused_kernel<4><<<RQ / 16, 256, 0, stream>>>(
      attn_sw + (size_t)2 * RQ * DI_, Wod_s, bod, nullptr,
      out + (size_t)RQ * DV_ + (size_t)RQ * DC_, nullptr, DI_, GEMM_F32, N_);
}
